// RaggedAttentionBlock_78898549227594
// MI455X (gfx1250) — matrix operand order repaired and checked by reading, not compiled, not run on hardware
//
#include <hip/hip_runtime.h>

// ---------------------------------------------------------------------------
// Types for CDNA5 WMMA (gfx1250, wave32)
// ---------------------------------------------------------------------------
typedef __attribute__((ext_vector_type(16))) __bf16 bf16x16;
typedef __attribute__((ext_vector_type(8)))  __bf16 bf16x8;
typedef __attribute__((ext_vector_type(8)))  float  f32x8;
typedef __attribute__((ext_vector_type(8)))  unsigned short u16x8;

#define DMODEL 768
#define NHEAD  12
#define DHEAD  64
#define DFF    3072
#define NQKV   2304

// DPP16 control codes
#define DPP_QUAD_XOR1   0xB1   // quad_perm(1,0,3,2)
#define DPP_QUAD_XOR2   0x4E   // quad_perm(2,3,0,1)
#define DPP_HALF_MIRROR 0x141  // reverse within 8 (== xor7)
#define DPP_ROW_MIRROR  0x140  // reverse within 16 (== xor15)

// float -> bf16 (round to nearest even), raw u16 storage
__device__ __forceinline__ unsigned short f2bf(float f) {
    unsigned u = __float_as_uint(f);
    u += 0x7FFFu + ((u >> 16) & 1u);
    return (unsigned short)(u >> 16);
}

__device__ __forceinline__ bf16x8 ld8(const unsigned short* p) {
    return *(const bf16x8*)p;   // 16B aligned by construction
}

__device__ __forceinline__ bf16x16 cmb(bf16x8 lo, bf16x8 hi) {
    bf16x16 r;
#pragma unroll
    for (int i = 0; i < 8; ++i) { r[i] = lo[i]; r[i + 8] = hi[i]; }
    return r;
}

// max(v, dpp_perm(v)) via DPP16 — pure VALU, no DS traffic, no waits.
template <int CTRL>
__device__ __forceinline__ float dppmax(float v) {
    int t = __builtin_amdgcn_update_dpp(0, __builtin_bit_cast(int, v), CTRL,
                                        0xF, 0xF, true);
    return fmaxf(v, __builtin_bit_cast(float, t));
}

// Width-16 max reduction over 8 independent rows using the DPP butterfly:
// xor1, xor2, (xor4 via half-mirror), (xor8 via row-mirror). The two 16-lane
// halves of the wave reduce independently, matching the WMMA C layout.
__device__ __forceinline__ void rowmax16_8(float a[8]) {
#pragma unroll
    for (int i = 0; i < 8; ++i) a[i] = dppmax<DPP_QUAD_XOR1>(a[i]);
#pragma unroll
    for (int i = 0; i < 8; ++i) a[i] = dppmax<DPP_QUAD_XOR2>(a[i]);
#pragma unroll
    for (int i = 0; i < 8; ++i) a[i] = dppmax<DPP_HALF_MIRROR>(a[i]);
#pragma unroll
    for (int i = 0; i < 8; ++i) a[i] = dppmax<DPP_ROW_MIRROR>(a[i]);
}

__device__ __forceinline__ float wsum32(float v) {
#pragma unroll
    for (int m = 16; m >= 1; m >>= 1) v += __shfl_xor(v, m, 32);
    return v;
}

// ---------------------------------------------------------------------------
// fp32 -> bf16 conversion kernel (weights)
// ---------------------------------------------------------------------------
__global__ void cvt_bf16_kernel(const float* __restrict__ in,
                                unsigned short* __restrict__ out, int n) {
    int i = blockIdx.x * 256 + threadIdx.x;
    if (i < n) out[i] = f2bf(in[i]);
}

// ---------------------------------------------------------------------------
// LayerNorm: one wave per token row (D=768 -> 24 elems/lane), bf16 output
// ---------------------------------------------------------------------------
__global__ __launch_bounds__(32) void ln_kernel(const float* __restrict__ x,
                                                const float* __restrict__ w,
                                                const float* __restrict__ b,
                                                unsigned short* __restrict__ out) {
    const int row  = blockIdx.x;
    const int lane = threadIdx.x & 31;
    const float* xr = x + (size_t)row * DMODEL;
    float v[24];
    float s = 0.f;
#pragma unroll
    for (int j = 0; j < 24; ++j) { v[j] = xr[j * 32 + lane]; s += v[j]; }
    const float mu = wsum32(s) * (1.f / DMODEL);
    float vs = 0.f;
#pragma unroll
    for (int j = 0; j < 24; ++j) { float d = v[j] - mu; vs += d * d; }
    const float rs = rsqrtf(wsum32(vs) * (1.f / DMODEL) + 1e-6f);
    unsigned short* orow = out + (size_t)row * DMODEL;
#pragma unroll
    for (int j = 0; j < 24; ++j) {
        int c = j * 32 + lane;
        orow[c] = f2bf((v[j] - mu) * rs * w[c] + b[c]);
    }
}

// ---------------------------------------------------------------------------
// Generic bf16 WMMA GEMM: C = A(MxK) * B(NxK)^T, wave tile 64x64.
// EP 0: qkv split epilogue  EP 1: resid + gamma*(acc+bias)  EP 2: GELU->bf16
// ---------------------------------------------------------------------------
template <int EP>
__global__ __launch_bounds__(32) void gemm_wmma_kernel(
    const unsigned short* __restrict__ A,   // [M][K] bf16
    const unsigned short* __restrict__ Bw,  // [N][K] bf16
    int M, int N, int K,
    const float* __restrict__ bias,         // [N]
    const float* __restrict__ gamma,        // [N]      (EP1)
    const float* __restrict__ resid,        // [M][N]   (EP1)
    float* __restrict__ outF,               // [M][N]   (EP1)
    unsigned short* __restrict__ outB,      // [M][N]   (EP2)
    unsigned short* __restrict__ qOut,      // EP0
    unsigned short* __restrict__ kOut,      // EP0
    unsigned short* __restrict__ vTOut,     // EP0: [D][total]
    int total) {
    const int lane = threadIdx.x & 31;
    const int half = lane >> 4;
    const int l16  = lane & 15;
    const int n0   = blockIdx.x * 64;
    const int row0 = blockIdx.y * 64;

    f32x8 acc[4][4];
#pragma unroll
    for (int mi = 0; mi < 4; ++mi)
#pragma unroll
        for (int ni = 0; ni < 4; ++ni)
#pragma unroll
            for (int i = 0; i < 8; ++i) acc[mi][ni][i] = 0.f;

    for (int k0 = 0; k0 < K; k0 += 32) {
        if (k0 + 32 < K) {
            __builtin_prefetch(Bw + (size_t)(n0 + l16) * K + k0 + 32, 0, 1);
            __builtin_prefetch(A + (size_t)(row0 + l16) * K + k0 + 32, 0, 1);
        }
        bf16x16 af[4];
#pragma unroll
        for (int mi = 0; mi < 4; ++mi) {
            const unsigned short* p =
                A + (size_t)(row0 + mi * 16 + l16) * K + k0 + half * 16;
            af[mi] = cmb(ld8(p), ld8(p + 8));
        }
        bf16x16 bfr[4];
#pragma unroll
        for (int ni = 0; ni < 4; ++ni) {
            const unsigned short* p =
                Bw + (size_t)(n0 + ni * 16 + l16) * K + k0 + half * 16;
            bfr[ni] = cmb(ld8(p), ld8(p + 8));
        }
#pragma unroll
        for (int mi = 0; mi < 4; ++mi)
#pragma unroll
            for (int ni = 0; ni < 4; ++ni)
                acc[mi][ni] = __builtin_amdgcn_wmma_f32_16x16x32_bf16(
                    false, af[mi], false, bfr[ni], (short)0, acc[mi][ni],
                    false, false);
    }

#pragma unroll
    for (int mi = 0; mi < 4; ++mi)
#pragma unroll
        for (int ni = 0; ni < 4; ++ni)
#pragma unroll
            for (int i = 0; i < 8; ++i) {
                const int row = row0 + mi * 16 + half * 8 + i;
                const int col = n0 + ni * 16 + l16;
                float v = acc[mi][ni][i] + bias[col];
                if (EP == 0) {
                    if (col < DMODEL)
                        qOut[(size_t)row * DMODEL + col] = f2bf(v);
                    else if (col < 2 * DMODEL)
                        kOut[(size_t)row * DMODEL + (col - DMODEL)] = f2bf(v);
                    else
                        vTOut[(size_t)(col - 2 * DMODEL) * total + row] = f2bf(v);
                } else if (EP == 1) {
                    outF[(size_t)row * N + col] =
                        resid[(size_t)row * N + col] + gamma[col] * v;
                } else {  // EP == 2: exact GELU
                    float g = 0.5f * v * (1.f + erff(v * 0.70710678118f));
                    outB[(size_t)row * N + col] = f2bf(g);
                }
            }
}

// ---------------------------------------------------------------------------
// Flash attention: one wave per (16-query tile, head), 32 keys per iteration.
// Row max via DPP16 butterflies (VALU only); row sum via a P x ones WMMA.
// ---------------------------------------------------------------------------
__global__ __launch_bounds__(32) void flash_attn_kernel(
    const unsigned short* __restrict__ qb,   // [total][768] bf16
    const unsigned short* __restrict__ kb,   // [total][768] bf16
    const unsigned short* __restrict__ vT,   // [768][total] bf16
    const int* __restrict__ cu, int nseq, int total,
    unsigned short* __restrict__ attnOut) {  // [total][768] bf16
    const int lane = threadIdx.x & 31;
    const int half = lane >> 4;
    const int l16  = lane & 15;
    const int q0   = blockIdx.x * 16;
    const int h    = blockIdx.y;

    int b = 0;
    while (b + 1 < nseq && q0 >= cu[b + 1]) ++b;
    const int s0  = cu[b];
    const int len = cu[b + 1] - s0;

    // Q fragments (two K-chunks of 32 over DHEAD=64)
    bf16x16 aq[2];
#pragma unroll
    for (int c = 0; c < 2; ++c) {
        const unsigned short* p =
            qb + (size_t)(q0 + l16) * DMODEL + h * DHEAD + c * 32 + half * 16;
        aq[c] = cmb(ld8(p), ld8(p + 8));
    }

    f32x8 o[4], ol;
#pragma unroll
    for (int j = 0; j < 4; ++j)
#pragma unroll
        for (int i = 0; i < 8; ++i) o[j][i] = 0.f;
#pragma unroll
    for (int i = 0; i < 8; ++i) ol[i] = 0.f;
    float mrow[8];
#pragma unroll
    for (int i = 0; i < 8; ++i) mrow[i] = -1e30f;

    __shared__ __attribute__((aligned(64))) unsigned short pl[16 * 32];
    const u16x8 uo = {0x3F80, 0x3F80, 0x3F80, 0x3F80,
                      0x3F80, 0x3F80, 0x3F80, 0x3F80};  // bf16 1.0
    const bf16x8  ob   = __builtin_bit_cast(bf16x8, uo);
    const bf16x16 ones = cmb(ob, ob);

    const int nkt = (len + 31) >> 5;   // 32 keys per iteration
    for (int kt = 0; kt < nkt; ++kt) {
        const int key0 = s0 + kt * 32;
        // ---- scores for two 16-key tiles --------------------------------
        int kc0 = key0 + l16;        if (kc0 > total - 1) kc0 = total - 1;
        int kc1 = key0 + 16 + l16;   if (kc1 > total - 1) kc1 = total - 1;
        bf16x16 bk0[2], bk1[2];
#pragma unroll
        for (int c = 0; c < 2; ++c) {
            const unsigned short* p0 =
                kb + (size_t)kc0 * DMODEL + h * DHEAD + c * 32 + half * 16;
            const unsigned short* p1 =
                kb + (size_t)kc1 * DMODEL + h * DHEAD + c * 32 + half * 16;
            bk0[c] = cmb(ld8(p0), ld8(p0 + 8));
            bk1[c] = cmb(ld8(p1), ld8(p1 + 8));
        }
        f32x8 sA, sB;
#pragma unroll
        for (int i = 0; i < 8; ++i) { sA[i] = 0.f; sB[i] = 0.f; }
        sA = __builtin_amdgcn_wmma_f32_16x16x32_bf16(false, aq[0], false, bk0[0],
                                                     (short)0, sA, false, false);
        sA = __builtin_amdgcn_wmma_f32_16x16x32_bf16(false, aq[1], false, bk0[1],
                                                     (short)0, sA, false, false);
        sB = __builtin_amdgcn_wmma_f32_16x16x32_bf16(false, aq[0], false, bk1[0],
                                                     (short)0, sB, false, false);
        sB = __builtin_amdgcn_wmma_f32_16x16x32_bf16(false, aq[1], false, bk1[1],
                                                     (short)0, sB, false, false);
        // ---- masked online softmax --------------------------------------
        const float add0 = (kt * 32 + l16 < len) ? 0.f : -1e30f;
        const float add1 = (kt * 32 + 16 + l16 < len) ? 0.f : -1e30f;
        float a0[8], a1[8], red[8], alpha[8];
#pragma unroll
        for (int i = 0; i < 8; ++i) {
            a0[i]  = sA[i] * 0.125f + add0;
            a1[i]  = sB[i] * 0.125f + add1;
            red[i] = fmaxf(a0[i], a1[i]);
        }
        rowmax16_8(red);   // DPP16 butterfly, no DS ops
#pragma unroll
        for (int i = 0; i < 8; ++i) {
            float mnew = fmaxf(mrow[i], red[i]);
            alpha[i] = __expf(mrow[i] - mnew);
            mrow[i]  = mnew;
            a0[i]    = __expf(a0[i] - mnew);
            a1[i]    = __expf(a1[i] - mnew);
        }
#pragma unroll
        for (int i = 0; i < 8; ++i) {
            ol[i] *= alpha[i];
#pragma unroll
            for (int j = 0; j < 4; ++j) o[j][i] *= alpha[i];
        }
        // ---- P (16x32) -> LDS -> full A fragment -------------------------
#pragma unroll
        for (int i = 0; i < 8; ++i) {
            pl[(i + half * 8) * 32 + l16]      = f2bf(a0[i]);
            pl[(i + half * 8) * 32 + 16 + l16] = f2bf(a1[i]);
        }
        asm volatile("s_wait_dscnt 0x0" ::: "memory");
        bf16x16 ap = cmb(ld8(pl + l16 * 32 + half * 16),
                         ld8(pl + l16 * 32 + 8 + half * 16));
        // row sums: P x ones (accumulated with same alpha rescaling)
        ol = __builtin_amdgcn_wmma_f32_16x16x32_bf16(false, ap, false, ones,
                                                     (short)0, ol, false, false);
        // ---- O += P x V --------------------------------------------------
        int krow = key0 + half * 16;
        if (krow > total - 16) krow = total - 16;  // masked lanes have P == 0
#pragma unroll
        for (int j = 0; j < 4; ++j) {
            const unsigned short* p =
                vT + (size_t)(h * DHEAD + j * 16 + l16) * total + krow;
            bf16x16 bv = cmb(ld8(p), ld8(p + 8));
            o[j] = __builtin_amdgcn_wmma_f32_16x16x32_bf16(
                false, ap, false, bv, (short)0, o[j], false, false);
        }
    }

    float inv[8];
#pragma unroll
    for (int i = 0; i < 8; ++i) inv[i] = 1.f / ol[i];
#pragma unroll
    for (int j = 0; j < 4; ++j)
#pragma unroll
        for (int i = 0; i < 8; ++i) {
            const int row = q0 + half * 8 + i;
            const int col = h * DHEAD + j * 16 + l16;
            attnOut[(size_t)row * DMODEL + col] = f2bf(o[j][i] * inv[i]);
        }
}

// ---------------------------------------------------------------------------
// Host-side launch
// ---------------------------------------------------------------------------
extern "C" void kernel_launch(void* const* d_in, const int* in_sizes, int n_in,
                              void* d_out, int out_size, void* d_ws,
                              size_t ws_size, hipStream_t stream) {
    const float* x     = (const float*)d_in[0];
    const float* n1w   = (const float*)d_in[1];
    const float* n1b   = (const float*)d_in[2];
    const float* qkv_w = (const float*)d_in[3];
    const float* qkv_b = (const float*)d_in[4];
    const float* pw    = (const float*)d_in[5];
    const float* pb    = (const float*)d_in[6];
    const float* ls1   = (const float*)d_in[7];
    const float* n2w   = (const float*)d_in[8];
    const float* n2b   = (const float*)d_in[9];
    const float* w1    = (const float*)d_in[10];
    const float* b1    = (const float*)d_in[11];
    const float* w2    = (const float*)d_in[12];
    const float* b2    = (const float*)d_in[13];
    const float* ls2   = (const float*)d_in[14];
    const int*   cu    = (const int*)d_in[15];

    const int total = in_sizes[0] / DMODEL;   // 4608
    const int nseq  = in_sizes[15] - 1;       // 8
    float* outp = (float*)d_out;

    char* ws = (char*)d_ws;
    size_t o = 0;
    auto take = [&](size_t bytes) {
        void* p = ws + o;
        o = (o + bytes + 255) & ~(size_t)255;
        return p;
    };
    unsigned short* xn   = (unsigned short*)take((size_t)total * DMODEL * 2);
    unsigned short* wqkv = (unsigned short*)take((size_t)NQKV * DMODEL * 2);
    unsigned short* wpr  = (unsigned short*)take((size_t)DMODEL * DMODEL * 2);
    unsigned short* wf1  = (unsigned short*)take((size_t)DFF * DMODEL * 2);
    unsigned short* wf2  = (unsigned short*)take((size_t)DMODEL * DFF * 2);
    unsigned short* qbuf = (unsigned short*)take((size_t)total * DMODEL * 2);
    unsigned short* kbuf = (unsigned short*)take((size_t)total * DMODEL * 2);
    unsigned short* vTb  = (unsigned short*)take((size_t)total * DMODEL * 2);
    unsigned short* attn = (unsigned short*)take((size_t)total * DMODEL * 2);
    float*          x1   = (float*)take((size_t)total * DMODEL * 4);
    unsigned short* xn2  = (unsigned short*)take((size_t)total * DMODEL * 2);
    unsigned short* hbuf = (unsigned short*)take((size_t)total * DFF * 2);

    // 1) weights fp32 -> bf16
    auto cvt = [&](const float* src, unsigned short* dst, int n) {
        cvt_bf16_kernel<<<(n + 255) / 256, 256, 0, stream>>>(src, dst, n);
    };
    cvt(qkv_w, wqkv, NQKV * DMODEL);
    cvt(pw, wpr, DMODEL * DMODEL);
    cvt(w1, wf1, DFF * DMODEL);
    cvt(w2, wf2, DMODEL * DFF);

    // 2) LN1
    ln_kernel<<<total, 32, 0, stream>>>(x, n1w, n1b, xn);

    // 3) QKV GEMM (M=total, N=2304, K=768) -> q, k, vT (bf16)
    gemm_wmma_kernel<0><<<dim3(NQKV / 64, total / 64), 32, 0, stream>>>(
        xn, wqkv, total, NQKV, DMODEL, qkv_b, nullptr, nullptr, nullptr,
        nullptr, qbuf, kbuf, vTb, total);

    // 4) flash attention
    flash_attn_kernel<<<dim3(total / 16, NHEAD), 32, 0, stream>>>(
        qbuf, kbuf, vTb, cu, nseq, total, attn);

    // 5) proj GEMM + residual + layer-scale -> x1 (fp32)
    gemm_wmma_kernel<1><<<dim3(DMODEL / 64, total / 64), 32, 0, stream>>>(
        attn, wpr, total, DMODEL, DMODEL, pb, ls1, x, x1, nullptr, nullptr,
        nullptr, nullptr, total);

    // 6) LN2
    ln_kernel<<<total, 32, 0, stream>>>(x1, n2w, n2b, xn2);

    // 7) FC1 GEMM + exact GELU -> h (bf16)
    gemm_wmma_kernel<2><<<dim3(DFF / 64, total / 64), 32, 0, stream>>>(
        xn2, wf1, total, DFF, DMODEL, b1, nullptr, nullptr, nullptr, hbuf,
        nullptr, nullptr, nullptr, total);

    // 8) FC2 GEMM + residual + layer-scale -> d_out (fp32)
    gemm_wmma_kernel<1><<<dim3(DMODEL / 64, total / 64), 32, 0, stream>>>(
        hbuf, wf2, total, DMODEL, DFF, b2, ls2, x1, outp, nullptr, nullptr,
        nullptr, nullptr, total);

    (void)n_in; (void)out_size; (void)ws_size;
}
